// GlobalSAModule_58600533786799
// MI455X (gfx1250) — matrix operand order repaired and checked by reading, not compiled, not run on hardware
//
#include <hip/hip_runtime.h>
#include <stdint.h>

typedef _Float16 half_t;
typedef __attribute__((ext_vector_type(16))) _Float16 v16h;
typedef __attribute__((ext_vector_type(8)))  _Float16 v8h;
typedef __attribute__((ext_vector_type(8)))  float    v8f;

#define BGRP 64
#define NPG  1024
#define NTOT (BGRP * NPG)
#define FDIM 256
#define KPAD 288            // 259 padded to 9 * 32
#define OUTD 1024
#define KNN  8

// ---------------------------------------------------------------------------
// Kernel 1: convert x (N x 256 fp32) into f16 activation matrix Xh (N x 288)
// ---------------------------------------------------------------------------
__global__ __launch_bounds__(256) void cvt_x_kernel(const float* __restrict__ x,
                                                    half_t* __restrict__ Xh) {
  int i = blockIdx.x * 256 + threadIdx.x;     // one float4 per thread
  float4 v = ((const float4*)x)[i];
  int row = i >> 6;                           // 64 float4 per row of 256
  int c   = (i & 63) * 4;
  half_t* p = Xh + (size_t)row * KPAD + c;
  p[0] = (half_t)v.x; p[1] = (half_t)v.y; p[2] = (half_t)v.z; p[3] = (half_t)v.w;
}

// ---------------------------------------------------------------------------
// Kernel 2: W (259 x 1024 fp32, row-major) -> Whc (f16, column-major 1024x288)
// ---------------------------------------------------------------------------
__global__ __launch_bounds__(256) void cvt_w_kernel(const float* __restrict__ W,
                                                    half_t* __restrict__ Whc) {
  int idx = blockIdx.x * 256 + threadIdx.x;   // over 1024*288
  int j = idx / KPAD;
  int k = idx - j * KPAD;
  float v = (k < FDIM + 3) ? W[(size_t)k * OUTD + j] : 0.0f;
  Whc[idx] = (half_t)v;
}

// ---------------------------------------------------------------------------
// Kernel 3: per-group centroid, rel, KNN(8), geometric messages.
// ---------------------------------------------------------------------------
__global__ __launch_bounds__(256) void knn_msg_kernel(const float* __restrict__ pos,
                                                      half_t* __restrict__ Xh) {
  __shared__ float relx[NPG], rely[NPG], relz[NPG], sq[NPG];
  __shared__ float red[3 * 256];

  const int g   = blockIdx.x;
  const int tid = threadIdx.x;
  const float* pg = pos + (size_t)g * NPG * 3;

  // centroid reduction
  float sx = 0.f, sy = 0.f, sz = 0.f;
  for (int p = tid; p < NPG; p += 256) {
    sx += pg[p * 3 + 0]; sy += pg[p * 3 + 1]; sz += pg[p * 3 + 2];
  }
  red[tid] = sx; red[256 + tid] = sy; red[512 + tid] = sz;
  __syncthreads();
  for (int s = 128; s > 0; s >>= 1) {
    if (tid < s) {
      red[tid]       += red[tid + s];
      red[256 + tid] += red[256 + tid + s];
      red[512 + tid] += red[512 + tid + s];
    }
    __syncthreads();
  }
  const float cx = red[0]   * (1.0f / NPG);
  const float cy = red[256] * (1.0f / NPG);
  const float cz = red[512] * (1.0f / NPG);

  for (int p = tid; p < NPG; p += 256) {
    float rx = pg[p * 3 + 0] - cx;
    float ry = pg[p * 3 + 1] - cy;
    float rz = pg[p * 3 + 2] - cz;
    relx[p] = rx; rely[p] = ry; relz[p] = rz;
    sq[p] = rx * rx + ry * ry + rz * rz;
  }
  __syncthreads();

  for (int p = tid; p < NPG; p += 256) {
    const float xi = relx[p], yi = rely[p], zi = relz[p], si = sq[p];
    float bd[KNN]; int bj[KNN];
#pragma unroll
    for (int q = 0; q < KNN; ++q) { bd[q] = 3.0e38f; bj[q] = 0; }

    for (int j = 0; j < NPG; ++j) {
      float dt = xi * relx[j] + yi * rely[j] + zi * relz[j];
      float d2 = si + sq[j] - 2.0f * dt;
      if (d2 < bd[KNN - 1]) {           // ripple insert (registers, fully unrolled)
        float cd = d2; int cj = j;
#pragma unroll
        for (int q = 0; q < KNN; ++q) {
          bool lt = cd < bd[q];
          float td = bd[q]; int tj = bj[q];
          bd[q] = lt ? cd : td; bj[q] = lt ? cj : tj;
          cd    = lt ? td : cd; cj    = lt ? tj : cj;
        }
      }
    }

    float ncs = 0.f, dts = 0.f;
#pragma unroll
    for (int q = 0; q < KNN; ++q) {
      int j = bj[q];
      float dt = xi * relx[j] + yi * rely[j] + zi * relz[j];
      float c2 = si * sq[j] - dt * dt;          // |v1 x v2|^2
      ncs += sqrtf(fmaxf(c2, 0.0f));
      dts += dt;
    }

    half_t* row = Xh + ((size_t)g * NPG + p) * KPAD;
    row[FDIM + 0] = (half_t)sqrtf(si);
    row[FDIM + 1] = (half_t)(ncs * (1.0f / KNN));
    row[FDIM + 2] = (half_t)(dts * (1.0f / KNN));
#pragma unroll
    for (int k = FDIM + 3; k < KPAD; ++k) row[k] = (half_t)0.0f;
  }
}

// ---------------------------------------------------------------------------
// Kernel 4: fused f16 WMMA GEMM + bias + ReLU + per-group max pooling.
// grid = (8 col-tiles, 64 groups), 256 threads = 8 waves.
// A streamed via double-buffered global_load_async_to_lds_b128 (ASYNCcnt).
// ---------------------------------------------------------------------------
__global__ __launch_bounds__(256) void gemm_pool_kernel(const half_t* __restrict__ Xh,
                                                        const half_t* __restrict__ Whc,
                                                        const float* __restrict__ bias,
                                                        float* __restrict__ out) {
  __shared__ __align__(16) half_t tileA[2][64 * KPAD];   // 2 x 36 KB

  const int g    = blockIdx.y;
  const int ct   = blockIdx.x;
  const int tid  = threadIdx.x;
  const int wave = tid >> 5;
  const int lane = tid & 31;
  const int l15  = lane & 15;
  const int hi   = lane >> 4;
  const int col  = ct * 128 + wave * 16 + l15;

  // Preload all 9 B k-blocks into registers.
  v16h breg[9];
  const half_t* wcol = Whc + (size_t)col * KPAD;
#pragma unroll
  for (int kb = 0; kb < 9; ++kb)
    breg[kb] = *(const v16h*)(wcol + kb * 32 + hi * 16);

  v8f runmax;
#pragma unroll
  for (int i = 0; i < 8; ++i) runmax[i] = -3.0e38f;

  const uint32_t ldsBase = (uint32_t)(uintptr_t)(&tileA[0][0]);
  const half_t* gsrc = Xh + (size_t)g * NPG * KPAD;

  // Issue one chunk (64 rows x 288 f16 = 36 KB) as 9 async b128 transfers
  // per thread into LDS buffer `buf` (direct global -> LDS, no VGPR staging).
  auto issue_chunk = [&](int chunk, int buf) {
    const uint64_t src = (uint64_t)(uintptr_t)(gsrc + (size_t)chunk * 64 * KPAD);
    const uint32_t dbase = ldsBase + (uint32_t)buf * (64 * KPAD * 2);
#pragma unroll
    for (int i = 0; i < 9; ++i) {
      const uint32_t v = (uint32_t)(tid + 256 * i);
      uint32_t lds = dbase + v * 16u;
      uint64_t ga  = src + (uint64_t)v * 16ull;
      asm volatile("global_load_async_to_lds_b128 %0, %1, off"
                   :: "v"(lds), "v"(ga) : "memory");
    }
  };

  issue_chunk(0, 0);
  for (int chunk = 0; chunk < 16; ++chunk) {
    const int buf = chunk & 1;
    asm volatile("s_wait_asynccnt 0x0" ::: "memory");   // this wave's loads for buf done
    __syncthreads();                                    // all waves' loads done; prev compute done
    if (chunk + 1 < 16) issue_chunk(chunk + 1, buf ^ 1);

#pragma unroll
    for (int sub = 0; sub < 4; ++sub) {
      const half_t* arow = &tileA[buf][(sub * 16 + l15) * KPAD];
      v8f c = {};
#pragma unroll
      for (int kb = 0; kb < 9; ++kb) {
        const int k0 = kb * 32 + hi * 16;
        v8h lo = *(const v8h*)(arow + k0);
        v8h hv = *(const v8h*)(arow + k0 + 8);
        v16h a = __builtin_shufflevector(lo, hv,
                                         0, 1, 2, 3, 4, 5, 6, 7,
                                         8, 9, 10, 11, 12, 13, 14, 15);
        c = __builtin_amdgcn_wmma_f32_16x16x32_f16(
            /*neg_a=*/false, a, /*neg_b=*/false, breg[kb],
            /*c_mod=*/(short)0, c, /*reuse_a=*/false, /*reuse_b=*/false);
      }
#pragma unroll
      for (int i = 0; i < 8; ++i) runmax[i] = fmaxf(runmax[i], c[i]);
    }
  }

  // Reduce: 8 VGPRs (rows M=i and M=i+8) then lane L vs L+16 (same column).
  float m = runmax[0];
#pragma unroll
  for (int i = 1; i < 8; ++i) m = fmaxf(m, runmax[i]);
  m = fmaxf(m, __shfl_xor(m, 16, 32));
  m = fmaxf(m + bias[col], 0.0f);
  if (hi == 0) out[(size_t)g * OUTD + col] = m;
}

// ---------------------------------------------------------------------------
// Kernel 5: tail outputs: pos_out = zeros(64,3), batch_out = arange(64)
// ---------------------------------------------------------------------------
__global__ void tail_kernel(float* __restrict__ out) {
  int t = threadIdx.x;
  if (t < BGRP * 3) out[BGRP * OUTD + t] = 0.0f;
  else if (t < BGRP * 3 + BGRP) {
    int i = t - BGRP * 3;
    ((int*)out)[BGRP * OUTD + BGRP * 3 + i] = i;
  }
}

// ---------------------------------------------------------------------------
extern "C" void kernel_launch(void* const* d_in, const int* in_sizes, int n_in,
                              void* d_out, int out_size, void* d_ws, size_t ws_size,
                              hipStream_t stream) {
  const float* x    = (const float*)d_in[0];
  const float* pos  = (const float*)d_in[1];
  const float* W    = (const float*)d_in[2];
  const float* bias = (const float*)d_in[3];
  // d_in[4] (batch) unused: groups are contiguous blocks of 1024 points.

  const size_t XH_BYTES = (size_t)NTOT * KPAD * sizeof(half_t);   // 36 MB
  half_t* Xh  = (half_t*)d_ws;
  half_t* Whc = (half_t*)((char*)d_ws + XH_BYTES);                // 576 KB
  float* out  = (float*)d_out;

  cvt_x_kernel<<<(NTOT * FDIM / 4) / 256, 256, 0, stream>>>(x, Xh);
  cvt_w_kernel<<<(OUTD * KPAD) / 256, 256, 0, stream>>>(W, Whc);
  knn_msg_kernel<<<BGRP, 256, 0, stream>>>(pos, Xh);
  gemm_pool_kernel<<<dim3(8, BGRP), 256, 0, stream>>>(Xh, Whc, bias, out);
  tail_kernel<<<1, 256, 0, stream>>>(out);
}
